// GraphTransformerModel_40776419508687
// MI455X (gfx1250) — matrix operand order repaired and checked by reading, not compiled, not run on hardware
//
#include <hip/hip_runtime.h>
#include <hip/hip_bf16.h>

// ---------------------------------------------------------------------------
// Graph Transformer forward, gfx1250 (CDNA5, wave32, WMMA f32_16x16x32_f16)
// One workgroup (256 thr = 8 waves) per graph; activations + staged f16
// weight panels all resident in LDS (305 KB of the 320 KB per WGP).
// Next-panel global_prefetch_b8 overlaps weight fetch with WMMA compute.
// ---------------------------------------------------------------------------

typedef __attribute__((ext_vector_type(16))) _Float16 v16h;
typedef __attribute__((ext_vector_type(8)))  float    v8f;

#define Gn     256
#define Nn     255
#define NNn    256
#define Hh     128
#define Ll     4
#define LDH    136   // fp32 h stride   (544B rows: 16B aligned)
#define LDQ    40    // f16 q/v stride  (80B rows:  16B aligned)
#define LDKT   264   // f16 kT stride   (528B rows: 16B aligned)
#define LDO    136   // f16 attn-out    (272B rows: 16B aligned)
#define LDSTG  40    // per-wave 16x32 staging stride
#define LDW1   40    // staged W panel strides
#define LDW2   136

#define WMMA(a,b,c) __builtin_amdgcn_wmma_f32_16x16x32_f16(false,(a),false,(b),(short)0,(c),false,false)

// A fragment (16x32 f16) from row-major fp32 LDS tile, leading dim ld.
// M = lane&15 ; lower half lanes hold K {0..7,16..23}, upper {8..15,24..31}.
__device__ __forceinline__ v16h ldsA_f32(const float* p, int ld) {
  const int lane = threadIdx.x & 31;
  const int kh = (lane >> 4) << 3;
  const float* r = p + (lane & 15) * ld;
  v16h a;
#pragma unroll
  for (int i = 0; i < 8; ++i) a[i] = (_Float16)r[kh + i];
#pragma unroll
  for (int i = 0; i < 8; ++i) a[8 + i] = (_Float16)r[16 + kh + i];
  return a;
}

__device__ __forceinline__ v16h ldsA_f16(const _Float16* p, int ld) {
  const int lane = threadIdx.x & 31;
  const int kh = (lane >> 4) << 3;
  const _Float16* r = p + (lane & 15) * ld;
  v16h a;
#pragma unroll
  for (int i = 0; i < 8; ++i) a[i] = r[kh + i];
#pragma unroll
  for (int i = 0; i < 8; ++i) a[8 + i] = r[16 + kh + i];
  return a;
}

// B fragment (32x16 f16) from f16 LDS: N = lane&15 ; lower half lanes hold K {0..7,16..23}, upper {8..15,24..31}.
__device__ __forceinline__ v16h ldB_f16lds(const _Float16* p, int ld) {
  const int lane = threadIdx.x & 31;
  const int kh = (lane >> 4) << 3;
  const _Float16* r = p + (lane & 15);
  v16h b;
#pragma unroll
  for (int i = 0; i < 8; ++i) b[i] = r[(kh + i) * ld];
#pragma unroll
  for (int i = 0; i < 8; ++i) b[8 + i] = r[(16 + kh + i) * ld];
  return b;
}

__device__ __forceinline__ float gelu_exact(float x) {
  return 0.5f * x * (1.0f + erff(x * 0.70710678118654752f));
}

__global__ __launch_bounds__(256, 1)
void graph_transformer_kernel(
    const float* __restrict__ x,    const float* __restrict__ Aadj,
    const float* __restrict__ We,   const float* __restrict__ be,
    const float* __restrict__ ctok,
    const float* __restrict__ Wq,   const float* __restrict__ bq,
    const float* __restrict__ Wk,   const float* __restrict__ bk,
    const float* __restrict__ Wv,   const float* __restrict__ bv,
    const float* __restrict__ Wo,   const float* __restrict__ bo,
    const float* __restrict__ ln1s, const float* __restrict__ ln1b,
    const float* __restrict__ W1,   const float* __restrict__ b1,
    const float* __restrict__ W2,   const float* __restrict__ b2,
    const float* __restrict__ ln2s, const float* __restrict__ ln2b,
    const float* __restrict__ Wh1,  const float* __restrict__ bh1,
    const float* __restrict__ Wh2,  const float* __restrict__ bh2,
    const float* __restrict__ Wh3,  const float* __restrict__ bh3,
    float* __restrict__ out)
{
  __shared__ float    shH [NNn * LDH];        // fp32 activations (residual/LN anchor)
  __shared__ _Float16 shQ [NNn * LDQ];        // per-head q (scale folded)
  __shared__ _Float16 shV [NNn * LDQ];        // per-head v
  __shared__ _Float16 shKT[32  * LDKT];       // per-head k transposed (dh x tokens)
  __shared__ _Float16 shO [NNn * LDO];        // attention output (f16, A-operand)
  __shared__ _Float16 shStg[8 * 16 * LDSTG];  // per-wave 16x32 staging
  __shared__ union {                          // staged f16 weight panels
    _Float16 qkv[3 * 128 * LDW1];             //  q/k/v head slices (128x32 each)
    _Float16 wo [128 * LDW2];                 //  Wo full (128x128)
    struct { _Float16 w1[128 * LDW1];         //  W1 col-slice (128x32)
             _Float16 w2[32  * LDW2]; } ffn;  //  W2 row-slice (32x128)
  } shW;
  __shared__ float z1[128];
  __shared__ float z2[64];

  const int g    = blockIdx.x;
  const int tid  = threadIdx.x;
  const int wave = __builtin_amdgcn_readfirstlane(tid >> 5);
  const int lane = tid & 31;
  const int col  = lane & 15;
  const int hi   = lane >> 4;
  const float scale = 0.17677669529663687f;   // 1/sqrt(32)
  const float* Ag = Aadj + (size_t)g * NNn * NNn;

  // ---------------- Phase 0: embedding + class token -----------------------
  {
    float xr[16];
    if (tid < Nn) {
#pragma unroll
      for (int k = 0; k < 16; ++k) xr[k] = x[(size_t)(g * Nn + tid) * 16 + k];
    }
    for (int c = 0; c < Hh; ++c) {
      float acc;
      if (tid < Nn) {
        acc = be[c];
#pragma unroll
        for (int k = 0; k < 16; ++k) acc += xr[k] * We[k * Hh + c];
      } else {
        acc = ctok[c];
      }
      shH[tid * LDH + c] = acc;
    }
  }
  __syncthreads();

  // ---------------- Transformer layers --------------------------------------
  for (int l = 0; l < Ll; ++l) {
    const float* Wql = Wq + l * Hh * Hh;  const float* bql = bq + l * Hh;
    const float* Wkl = Wk + l * Hh * Hh;  const float* bkl = bk + l * Hh;
    const float* Wvl = Wv + l * Hh * Hh;  const float* bvl = bv + l * Hh;
    const float* Wol = Wo + l * Hh * Hh;  const float* bol = bo + l * Hh;
    const float* W1l = W1 + l * Hh * 2 * Hh; const float* b1l = b1 + l * 2 * Hh;
    const float* W2l = W2 + l * 2 * Hh * Hh; const float* b2l = b2 + l * Hh;

    // prefetch this wave's adjacency-mask block (32 rows x 1KB = 256 lines)
    {
      const char* Abase = (const char*)(Ag) + (size_t)wave * 32 * NNn * 4;
#pragma unroll
      for (int q = 0; q < 8; ++q)
        __builtin_prefetch(Abase + (q * 32 + lane) * 128, 0, 1);
    }

    // ---- attention, head by head ----
    for (int head = 0; head < 4; ++head) {
      // stage q/k/v head slices (128x32 each) as f16 panels
      for (int idx = tid; idx < 3 * 128 * 32; idx += 256) {
        const int m   = idx >> 12;
        const int rem = idx & 4095;
        const int row = rem >> 5, c = rem & 31;
        const float* W = (m == 0) ? Wql : (m == 1) ? Wkl : Wvl;
        shW.qkv[m * 128 * LDW1 + row * LDW1 + c] =
            (_Float16)W[row * Hh + head * 32 + c];
      }
      // prefetch next panel while this head computes
      if (head < 3) {
        for (int idx = tid; idx < 384; idx += 256) {
          const int m = idx >> 7, row = idx & 127;
          const float* W = (m == 0) ? Wql : (m == 1) ? Wkl : Wvl;
          __builtin_prefetch(&W[row * Hh + (head + 1) * 32], 0, 1);
        }
      } else {
        for (int idx = tid; idx < 512; idx += 256)
          __builtin_prefetch((const char*)Wol + idx * 128, 0, 1);
      }
      __syncthreads();

      // q/k/v projections for this head: 96 tile-jobs over 8 waves
      for (int job = wave; job < 96; job += 8) {
        const int mat   = job / 32;       // 0=q 1=k 2=v
        const int rem   = job % 32;
        const int mtile = rem >> 1;
        const int nt    = rem & 1;
        const float* bb = (mat == 0) ? bql : (mat == 1) ? bkl : bvl;
        const _Float16* Wp = &shW.qkv[mat * 128 * LDW1];
        v8f acc = {};
#pragma unroll
        for (int ks = 0; ks < 4; ++ks) {
          v16h a = ldsA_f32(&shH[mtile * 16 * LDH + ks * 32], LDH);
          v16h b = ldB_f16lds(&Wp[(ks * 32) * LDW1 + nt * 16], LDW1);
          acc = WMMA(a, b, acc);
        }
        const float bn = bb[head * 32 + nt * 16 + col];
        if (mat == 0) {
#pragma unroll
          for (int r = 0; r < 8; ++r) {
            const int row = mtile * 16 + r + 8 * hi;
            shQ[row * LDQ + nt * 16 + col] = (_Float16)((acc[r] + bn) * scale);
          }
        } else if (mat == 1) {
#pragma unroll
          for (int r = 0; r < 8; ++r) {
            const int row = mtile * 16 + r + 8 * hi;
            shKT[(nt * 16 + col) * LDKT + row] = (_Float16)(acc[r] + bn);
          }
        } else {
#pragma unroll
          for (int r = 0; r < 8; ++r) {
            const int row = mtile * 16 + r + 8 * hi;
            shV[row * LDQ + nt * 16 + col] = (_Float16)(acc[r] + bn);
          }
        }
      }
      __syncthreads();

      // masked softmax attention: each wave owns 2 row-tiles (16 rows each)
      for (int it = 0; it < 2; ++it) {
        const int mt = wave * 2 + it;
        v16h aq = ldsA_f16(&shQ[mt * 16 * LDQ], LDQ);
        v8f s[16];
#pragma unroll
        for (int t = 0; t < 16; ++t) {
          v16h b = ldB_f16lds(&shKT[t * 16], LDKT);
          v8f z = {};
          s[t] = WMMA(aq, b, z);
        }
        // adjacency mask + row max
        float mx[8], sum[8];
#pragma unroll
        for (int r = 0; r < 8; ++r) mx[r] = -3.0e38f;
#pragma unroll
        for (int t = 0; t < 16; ++t) {
#pragma unroll
          for (int r = 0; r < 8; ++r) {
            const int i = mt * 16 + r + 8 * hi;
            const int j = t * 16 + col;
            const float av = Ag[i * NNn + j];
            const float sv = (av > 0.0f) ? s[t][r] : -1.0e9f;
            s[t][r] = sv;
            mx[r] = fmaxf(mx[r], sv);
          }
        }
#pragma unroll
        for (int r = 0; r < 8; ++r) {
          float m = mx[r];
          m = fmaxf(m, __shfl_xor(m, 1, 32));
          m = fmaxf(m, __shfl_xor(m, 2, 32));
          m = fmaxf(m, __shfl_xor(m, 4, 32));
          m = fmaxf(m, __shfl_xor(m, 8, 32));
          mx[r] = m;
          sum[r] = 0.0f;
        }
#pragma unroll
        for (int t = 0; t < 16; ++t) {
#pragma unroll
          for (int r = 0; r < 8; ++r) {
            const float e = __expf(s[t][r] - mx[r]);
            s[t][r] = e;
            sum[r] += e;
          }
        }
#pragma unroll
        for (int r = 0; r < 8; ++r) {
          float sm = sum[r];
          sm += __shfl_xor(sm, 1, 32);
          sm += __shfl_xor(sm, 2, 32);
          sm += __shfl_xor(sm, 4, 32);
          sm += __shfl_xor(sm, 8, 32);
          sum[r] = sm;
        }
        // p @ v  (stage 16x32 prob chunks C-layout -> A-layout via LDS)
        _Float16* stg = &shStg[wave * 16 * LDSTG];
        v8f oa0 = {}, oa1 = {};
        for (int ks = 0; ks < 8; ++ks) {
#pragma unroll
          for (int tt = 0; tt < 2; ++tt) {
            const int t = ks * 2 + tt;
#pragma unroll
            for (int r = 0; r < 8; ++r)
              stg[(r + 8 * hi) * LDSTG + tt * 16 + col] = (_Float16)s[t][r];
          }
          asm volatile("s_wait_dscnt 0" ::: "memory");
          v16h ap  = ldsA_f16(stg, LDSTG);
          v16h bv0 = ldB_f16lds(&shV[(ks * 32) * LDQ + 0], LDQ);
          oa0 = WMMA(ap, bv0, oa0);
          v16h bv1 = ldB_f16lds(&shV[(ks * 32) * LDQ + 16], LDQ);
          oa1 = WMMA(ap, bv1, oa1);
        }
#pragma unroll
        for (int r = 0; r < 8; ++r) {
          const int row = mt * 16 + r + 8 * hi;
          const float inv = 1.0f / sum[r];
          shO[row * LDO + head * 32 + col]      = (_Float16)(oa0[r] * inv);
          shO[row * LDO + head * 32 + 16 + col] = (_Float16)(oa1[r] * inv);
        }
      }
      __syncthreads();
    } // head

    // ---- output projection + residual (h += o @ Wo + bo) ----
    for (int idx = tid; idx < Hh * Hh; idx += 256) {
      const int row = idx >> 7, c = idx & 127;
      shW.wo[row * LDW2 + c] = (_Float16)Wol[idx];
    }
    // prefetch first FFN panel
    if (tid < 128) __builtin_prefetch(&W1l[tid * 256], 0, 1);
    else           __builtin_prefetch((const char*)W2l + (tid - 128) * 128, 0, 1);
    __syncthreads();
    for (int job = wave; job < 128; job += 8) {
      const int mtile = job >> 3;
      const int n0    = (job & 7) * 16;
      v8f acc = {};
#pragma unroll
      for (int ks = 0; ks < 4; ++ks) {
        v16h a = ldsA_f16(&shO[mtile * 16 * LDO + ks * 32], LDO);
        v16h b = ldB_f16lds(&shW.wo[(ks * 32) * LDW2 + n0], LDW2);
        acc = WMMA(a, b, acc);
      }
      const float bn = bol[n0 + col];
#pragma unroll
      for (int r = 0; r < 8; ++r) {
        const int row = mtile * 16 + r + 8 * hi;
        shH[row * LDH + n0 + col] += acc[r] + bn;
      }
    }
    __syncthreads();

    // ---- LayerNorm 1 (one row per thread) ----
    {
      float* hr = &shH[tid * LDH];
      float m = 0.0f;
      for (int c = 0; c < Hh; ++c) m += hr[c];
      m *= (1.0f / Hh);
      float v = 0.0f;
      for (int c = 0; c < Hh; ++c) { const float d = hr[c] - m; v += d * d; }
      v *= (1.0f / Hh);
      const float rinv = rsqrtf(v + 1e-5f);
      const float* sc = ln1s + l * Hh;
      const float* bc = ln1b + l * Hh;
      for (int c = 0; c < Hh; ++c) hr[c] = (hr[c] - m) * rinv * sc[c] + bc[c];
    }
    __syncthreads();

    // ---- FFN: h += gelu(h@W1+b1)@W2 + b2, K-chunked with staged panels ----
    {
      v8f facc[2][8] = {};
      for (int ks = 0; ks < 8; ++ks) {
        // stage W1[:, ks*32:+32] and W2[ks*32:+32, :] as f16
        for (int idx = tid; idx < 4096; idx += 256) {
          const int row = idx >> 5, c = idx & 31;
          shW.ffn.w1[row * LDW1 + c] = (_Float16)W1l[row * (2 * Hh) + ks * 32 + c];
        }
        for (int idx = tid; idx < 4096; idx += 256) {
          const int r32 = idx >> 7, c = idx & 127;
          shW.ffn.w2[r32 * LDW2 + c] = (_Float16)W2l[(ks * 32 + r32) * Hh + c];
        }
        // prefetch next panel while this chunk computes
        if (ks < 7) {
          if (tid < 128)
            __builtin_prefetch(&W1l[tid * 256 + (ks + 1) * 32], 0, 1);
          else
            __builtin_prefetch((const char*)(W2l + (ks + 1) * 32 * Hh) + (tid - 128) * 128, 0, 1);
        }
        __syncthreads();
        for (int it = 0; it < 2; ++it) {
          const int mt = wave * 2 + it;
          _Float16* stg = &shStg[wave * 16 * LDSTG];
#pragma unroll
          for (int tt = 0; tt < 2; ++tt) {
            v8f acc = {};
#pragma unroll
            for (int k2 = 0; k2 < 4; ++k2) {
              v16h a = ldsA_f32(&shH[mt * 16 * LDH + k2 * 32], LDH);
              v16h b = ldB_f16lds(&shW.ffn.w1[(k2 * 32) * LDW1 + tt * 16], LDW1);
              acc = WMMA(a, b, acc);
            }
            const float bn = b1l[ks * 32 + tt * 16 + col];
#pragma unroll
            for (int r = 0; r < 8; ++r) {
              const float xv = acc[r] + bn;
              stg[(r + 8 * hi) * LDSTG + tt * 16 + col] = (_Float16)gelu_exact(xv);
            }
          }
          asm volatile("s_wait_dscnt 0" ::: "memory");
          v16h ap = ldsA_f16(stg, LDSTG);
#pragma unroll
          for (int nt2 = 0; nt2 < 8; ++nt2) {
            v16h b = ldB_f16lds(&shW.ffn.w2[nt2 * 16], LDW2);
            facc[it][nt2] = WMMA(ap, b, facc[it][nt2]);
          }
        }
        __syncthreads();
      }
      // write back residual
#pragma unroll
      for (int it = 0; it < 2; ++it) {
        const int mt = wave * 2 + it;
#pragma unroll
        for (int nt2 = 0; nt2 < 8; ++nt2) {
          const float bn = b2l[nt2 * 16 + col];
#pragma unroll
          for (int r = 0; r < 8; ++r) {
            const int row = mt * 16 + r + 8 * hi;
            shH[row * LDH + nt2 * 16 + col] += facc[it][nt2][r] + bn;
          }
        }
      }
    }
    __syncthreads();

    // ---- LayerNorm 2 ----
    {
      float* hr = &shH[tid * LDH];
      float m = 0.0f;
      for (int c = 0; c < Hh; ++c) m += hr[c];
      m *= (1.0f / Hh);
      float v = 0.0f;
      for (int c = 0; c < Hh; ++c) { const float d = hr[c] - m; v += d * d; }
      v *= (1.0f / Hh);
      const float rinv = rsqrtf(v + 1e-5f);
      const float* sc = ln2s + l * Hh;
      const float* bc = ln2b + l * Hh;
      for (int c = 0; c < Hh; ++c) hr[c] = (hr[c] - m) * rinv * sc[c] + bc[c];
    }
    __syncthreads();
  } // layer

  // ---------------- Head MLP on class token (row 255) -----------------------
  if (tid < 128) {
    float acc = bh1[tid];
    for (int c = 0; c < 128; ++c) acc += shH[Nn * LDH + c] * Wh1[c * 128 + tid];
    z1[tid] = gelu_exact(acc);
  }
  __syncthreads();
  if (tid < 64) {
    float acc = bh2[tid];
    for (int c = 0; c < 128; ++c) acc += z1[c] * Wh2[c * 64 + tid];
    z2[tid] = gelu_exact(acc);
  }
  __syncthreads();
  if (tid < 10) {
    float acc = bh3[tid];
    for (int c = 0; c < 64; ++c) acc += z2[c] * Wh3[c * 10 + tid];
    out[g * 10 + tid] = acc;
  }
}

extern "C" void kernel_launch(void* const* d_in, const int* in_sizes, int n_in,
                              void* d_out, int out_size, void* d_ws, size_t ws_size,
                              hipStream_t stream) {
  (void)in_sizes; (void)n_in; (void)out_size; (void)d_ws; (void)ws_size;
  const float* x    = (const float*)d_in[0];
  const float* Aadj = (const float*)d_in[1];
  const float* We   = (const float*)d_in[2];
  const float* be   = (const float*)d_in[3];
  const float* ctok = (const float*)d_in[4];
  const float* Wq   = (const float*)d_in[5];
  const float* bq   = (const float*)d_in[6];
  const float* Wk   = (const float*)d_in[7];
  const float* bk   = (const float*)d_in[8];
  const float* Wv   = (const float*)d_in[9];
  const float* bv   = (const float*)d_in[10];
  const float* Wo   = (const float*)d_in[11];
  const float* bo   = (const float*)d_in[12];
  const float* ln1s = (const float*)d_in[13];
  const float* ln1b = (const float*)d_in[14];
  const float* W1   = (const float*)d_in[15];
  const float* b1   = (const float*)d_in[16];
  const float* W2   = (const float*)d_in[17];
  const float* b2   = (const float*)d_in[18];
  const float* ln2s = (const float*)d_in[19];
  const float* ln2b = (const float*)d_in[20];
  const float* Wh1  = (const float*)d_in[21];
  const float* bh1  = (const float*)d_in[22];
  const float* Wh2  = (const float*)d_in[23];
  const float* bh2  = (const float*)d_in[24];
  const float* Wh3  = (const float*)d_in[25];
  const float* bh3  = (const float*)d_in[26];
  float* out = (float*)d_out;

  graph_transformer_kernel<<<dim3(Gn), dim3(256), 0, stream>>>(
      x, Aadj, We, be, ctok, Wq, bq, Wk, bk, Wv, bv, Wo, bo,
      ln1s, ln1b, W1, b1, W2, b2, ln2s, ln2b,
      Wh1, bh1, Wh2, bh2, Wh3, bh3, out);
}
